// CausalSelfAttention_46179488366618
// MI455X (gfx1250) — matrix operand order repaired and checked by reading, not compiled, not run on hardware
//
#include <hip/hip_runtime.h>
#include <hip/hip_bf16.h>

// ---- problem constants (match reference) ----
#define BATCH 4
#define SLEN  2048
#define DDIM  1024
#define NH    16
#define HDIM  64
#define MROWS (BATCH * SLEN)   // 8192
#define NQKV  (3 * DDIM)       // 3072

typedef __attribute__((ext_vector_type(16))) __bf16        v16bf;
typedef __attribute__((ext_vector_type(8)))  float         v8f;
typedef __attribute__((ext_vector_type(4)))  unsigned int  u32x4;

union FragAB { u32x4 u[2]; v16bf v; };

__device__ __forceinline__ v8f vzero8() {
  v8f z = {0.f, 0.f, 0.f, 0.f, 0.f, 0.f, 0.f, 0.f};
  return z;
}

__device__ __forceinline__ __bf16 f32_to_bf16(float f) {
  union { float f; unsigned u; } c; c.f = f;
  unsigned u = c.u;
  unsigned r = u + 0x7fffu + ((u >> 16) & 1u);   // round-to-nearest-even
  union { unsigned short s; __bf16 b; } o; o.s = (unsigned short)(r >> 16);
  return o.b;
}

// Generic (flat) shared-memory address -> LDS byte offset.
// ISA 10.2: inside the LDS aperture, the LDS offset is the low 32 bits of the address.
__device__ __forceinline__ unsigned lds_off(const void* p) {
  return (unsigned)(size_t)p;
}

// One per-lane 16B async Global->LDS transfer (ASYNCcnt-tracked).
__device__ __forceinline__ void async_copy16(unsigned ldsByteOff, const void* gaddr) {
  asm volatile("global_load_async_to_lds_b128 %0, %1, off"
               :: "v"(ldsByteOff), "v"(gaddr) : "memory");
}
// Async loads retire in order (ISA 08 §4.1): waiting asynccnt<=6 after issuing
// 6 new transfers guarantees the previous chunk's 6 transfers completed.
__device__ __forceinline__ void wait_async(bool pending6) {
  if (pending6) asm volatile("s_wait_asynccnt 0x6" ::: "memory");
  else          asm volatile("s_wait_asynccnt 0x0" ::: "memory");
}

// A fragment: 16x32 bf16 (MxK) from row-major src (leading dim = ld elements).
// Layout (ISA 7.12.2): lanes 0-15 -> M=lane, K = {0..7, 16..23};
//                      lanes 16-31 -> M=lane-16, K = {8..15, 24..31}.
__device__ __forceinline__ v16bf load_frag_a(const __bf16* base, int ld, int lane) {
  int m    = lane & 15;
  int koff = (lane < 16) ? 0 : 8;
  const __bf16* p = base + (size_t)m * ld + koff;
  FragAB f;
  f.u[0] = *(const u32x4*)(p);        // K koff..koff+7
  f.u[1] = *(const u32x4*)(p + 16);   // K koff+16..koff+23
  return f.v;
}

// B fragment: 32x16 bf16 (KxN) from N-major storage (row n holds all K
// contiguously, leading dim = ld). Layout: lanes 0-15 hold K={0..7,16..23} of
// column N=lane, lanes 16-31 hold K={8..15,24..31} of column N=lane-16.
__device__ __forceinline__ v16bf load_frag_b(const __bf16* baseT, int ld, int lane) {
  int n   = lane & 15;
  int kof = (lane < 16) ? 0 : 8;
  const __bf16* p = baseT + (size_t)n * ld + kof;
  FragAB f;
  f.u[0] = *(const u32x4*)(p);       // K kof..kof+7
  f.u[1] = *(const u32x4*)(p + 16);  // K kof+16..kof+23
  return f.v;
}

#define WMMA_BF16(a, b, c) \
  __builtin_amdgcn_wmma_f32_16x16x32_bf16(false, (a), false, (b), (short)0, (c), false, false)

// Stage one 128x64 A tile + 64x64 B tile (bf16) into LDS: 6 async 16B
// transfers per thread (4 for A, 2 for B).
__device__ __forceinline__ void stage_tiles(const __bf16* __restrict__ A,
                                            const __bf16* __restrict__ Bt,
                                            int mblk, int n0, int kc,
                                            __bf16 (*Ast)[64], __bf16 (*Bst)[64],
                                            int tid) {
#pragma unroll
  for (int i = 0; i < 4; ++i) {
    int t = tid + i * 256, row = t >> 3, c8 = t & 7;
    async_copy16(lds_off(&Ast[row][c8 * 8]),
                 A + (size_t)(mblk + row) * DDIM + kc + c8 * 8);
  }
#pragma unroll
  for (int i = 0; i < 2; ++i) {
    int t = tid + i * 256, row = t >> 3, c8 = t & 7;
    async_copy16(lds_off(&Bst[row][c8 * 8]),
                 Bt + (size_t)(n0 + row) * DDIM + kc + c8 * 8);
  }
}

// 2 k-steps x 4 WMMAs from the staged tiles; fragments hoisted into distinct
// locals so the compiler can batch the 10 ds_load_b128s with partial waits.
__device__ __forceinline__ void compute_chunk(__bf16 (*Ast)[64], __bf16 (*Bst)[64],
                                              int wave, int lane, v8f acc[4]) {
#pragma unroll
  for (int ks = 0; ks < 64; ks += 32) {
    v16bf a  = load_frag_a(&Ast[wave * 16][0] + ks, 64, lane);
    v16bf b0 = load_frag_b(&Bst[0][0]  + ks, 64, lane);
    v16bf b1 = load_frag_b(&Bst[16][0] + ks, 64, lane);
    v16bf b2 = load_frag_b(&Bst[32][0] + ks, 64, lane);
    v16bf b3 = load_frag_b(&Bst[48][0] + ks, 64, lane);
    acc[0] = WMMA_BF16(a, b0, acc[0]);
    acc[1] = WMMA_BF16(a, b1, acc[1]);
    acc[2] = WMMA_BF16(a, b2, acc[2]);
    acc[3] = WMMA_BF16(a, b3, acc[3]);
  }
}

// ---------------- cast / transpose helpers ----------------
__global__ __launch_bounds__(256) void cast_f32_to_bf16(const float* __restrict__ in,
                                                        __bf16* __restrict__ out, int n) {
  int i = blockIdx.x * 256 + threadIdx.x;
  if (i < n) out[i] = f32_to_bf16(in[i]);
}

// in: [rows, cols] f32 row-major -> out: [cols, rows] bf16 (transposed)
__global__ __launch_bounds__(256) void transpose_cast(const float* __restrict__ in,
                                                      __bf16* __restrict__ out,
                                                      int rows, int cols) {
  int i = blockIdx.x * 256 + threadIdx.x;
  if (i < rows * cols) {
    int r = i / cols, c = i % cols;
    out[(size_t)c * rows + r] = f32_to_bf16(in[i]);
  }
}

// ---------------- QKV projection GEMM ----------------
// C[8192,3072] = X[8192,1024] @ Wqkv[1024,3072]; WT is [3072,1024] (N-major).
// Block = 8 waves -> 128x64 output tile. Double-buffered async Global->LDS
// staging: issue chunk kc+64 before computing chunk kc to hide HBM/L2 latency
// behind the 8 WMMAs per chunk.
// Output scattered: q -> [b,h,s,hd], k -> [b,h,s,hd], v -> [b,h,hd,s].
__global__ __launch_bounds__(256) void qkv_gemm_kernel(const __bf16* __restrict__ X,
                                                       const __bf16* __restrict__ WT,
                                                       __bf16* __restrict__ Q,
                                                       __bf16* __restrict__ K,
                                                       __bf16* __restrict__ VT) {
  __shared__ __align__(16) __bf16 Astage[2][128][64];   // 32 KB
  __shared__ __align__(16) __bf16 Bstage[2][64][64];    // 16 KB

  const int tid  = threadIdx.x;
  const int lane = tid & 31;
  const int wave = tid >> 5;
  const int n0   = blockIdx.x * 64;       // 0..3008
  const int mblk = blockIdx.y * 128;      // 0..8064
  const int m0   = mblk + wave * 16;

  v8f acc[4] = {vzero8(), vzero8(), vzero8(), vzero8()};

  stage_tiles(X, WT, mblk, n0, 0, Astage[0], Bstage[0], tid);
  int cur = 0;
  for (int kc = 0; kc < DDIM; kc += 64) {
    const bool haveNext = (kc + 64) < DDIM;
    if (haveNext)
      stage_tiles(X, WT, mblk, n0, kc + 64, Astage[cur ^ 1], Bstage[cur ^ 1], tid);
    wait_async(haveNext);
    __syncthreads();
    compute_chunk(Astage[cur], Bstage[cur], wave, lane, acc);
    __syncthreads();   // all reads done before this buffer is re-targeted
    cur ^= 1;
  }

#pragma unroll
  for (int j = 0; j < 4; ++j) {
    int n = n0 + j * 16 + (lane & 15);
#pragma unroll
    for (int r = 0; r < 8; ++r) {
      int m = m0 + r + ((lane < 16) ? 0 : 8);
      int b = m >> 11;          // / SLEN
      int s = m & (SLEN - 1);
      __bf16 val = f32_to_bf16(acc[j][r]);
      if (n < DDIM) {
        int h = n >> 6, hd = n & 63;
        Q[(((size_t)(b * NH + h)) * SLEN + s) * HDIM + hd] = val;
      } else if (n < 2 * DDIM) {
        int n2 = n - DDIM; int h = n2 >> 6, hd = n2 & 63;
        K[(((size_t)(b * NH + h)) * SLEN + s) * HDIM + hd] = val;
      } else {
        int n2 = n - 2 * DDIM; int h = n2 >> 6, hd = n2 & 63;
        VT[(((size_t)(b * NH + h)) * HDIM + hd) * SLEN + s] = val;
      }
    }
  }
}

// ---------------- flash attention ----------------
// Grid: (SLEN/16/8, BATCH*NH). One wave = one 16-row q tile, streams keys in
// 32-wide blocks with online softmax. Causal mask applied analytically.
// K/V fragment loads hit L2 (each bh's K/V is 256 KB bf16, reused by 128 waves).
__global__ __launch_bounds__(256) void attn_kernel(const __bf16* __restrict__ Q,
                                                   const __bf16* __restrict__ K,
                                                   const __bf16* __restrict__ VT,
                                                   __bf16* __restrict__ CTX) {
  __shared__ __align__(16) __bf16 Pstage[8][16 * 32];   // per-wave P tile (bf16)

  const int lane = threadIdx.x & 31;
  const int wave = threadIdx.x >> 5;
  const int qt   = blockIdx.x * 8 + wave;    // 0..127
  const int bh   = blockIdx.y;               // 0..63
  const int b    = bh >> 4, h = bh & 15;

  const __bf16* qbase = Q  + ((size_t)bh * SLEN + qt * 16) * HDIM;
  const __bf16* kbase = K  + (size_t)bh * SLEN * HDIM;
  const __bf16* vbase = VT + (size_t)bh * HDIM * SLEN;
  __bf16* pst = &Pstage[wave][0];

  // q A-fragments are loop-invariant: hd 0..31 and 32..63
  const v16bf aq0 = load_frag_a(qbase + 0,  HDIM, lane);
  const v16bf aq1 = load_frag_a(qbase + 32, HDIM, lane);

  float mrow[8], lrow[8];
#pragma unroll
  for (int r = 0; r < 8; ++r) { mrow[r] = -1e30f; lrow[r] = 0.f; }
  v8f o[4] = {vzero8(), vzero8(), vzero8(), vzero8()};

  const float scale = 0.125f;          // 1/sqrt(64)
  const int qhi = qt * 16 + 15;

  for (int k0 = 0; k0 <= qhi; k0 += 32) {
    // ---- scores: two 16x16 tiles over HD=64 (2 WMMAs each) ----
    v8f s0 = vzero8(), s1 = vzero8();
    s0 = WMMA_BF16(aq0, load_frag_b(kbase + (size_t)k0 * HDIM + 0,  HDIM, lane), s0);
    s0 = WMMA_BF16(aq1, load_frag_b(kbase + (size_t)k0 * HDIM + 32, HDIM, lane), s0);
    s1 = WMMA_BF16(aq0, load_frag_b(kbase + (size_t)(k0 + 16) * HDIM + 0,  HDIM, lane), s1);
    s1 = WMMA_BF16(aq1, load_frag_b(kbase + (size_t)(k0 + 16) * HDIM + 32, HDIM, lane), s1);

    const int kcol = lane & 15;
    const int mofs = (lane < 16) ? 0 : 8;
#pragma unroll
    for (int r = 0; r < 8; ++r) {
      int qi = qt * 16 + r + mofs;
      float e0 = (k0 + kcol      <= qi) ? s0[r] * scale : -1e30f;
      float e1 = (k0 + 16 + kcol <= qi) ? s1[r] * scale : -1e30f;
      // row max over the 16-lane half holding this row
      float loc = fmaxf(e0, e1);
      loc = fmaxf(loc, __shfl_xor(loc, 1, 32));
      loc = fmaxf(loc, __shfl_xor(loc, 2, 32));
      loc = fmaxf(loc, __shfl_xor(loc, 4, 32));
      loc = fmaxf(loc, __shfl_xor(loc, 8, 32));
      float mnew  = fmaxf(mrow[r], loc);
      float alpha = __expf(mrow[r] - mnew);
      float p0 = __expf(e0 - mnew);
      float p1 = __expf(e1 - mnew);
      float ps = p0 + p1;
      ps += __shfl_xor(ps, 1, 32);
      ps += __shfl_xor(ps, 2, 32);
      ps += __shfl_xor(ps, 4, 32);
      ps += __shfl_xor(ps, 8, 32);
      lrow[r] = lrow[r] * alpha + ps;
      mrow[r] = mnew;
      o[0][r] *= alpha; o[1][r] *= alpha; o[2][r] *= alpha; o[3][r] *= alpha;
      // stage P (C layout -> LDS row-major 16x32)
      int mr = r + mofs;
      pst[mr * 32 + kcol]      = f32_to_bf16(p0);
      pst[mr * 32 + 16 + kcol] = f32_to_bf16(p1);
    }
    // same-wave LDS store -> load ordering
    asm volatile("s_wait_dscnt 0" ::: "memory");

    // ---- P (16x32) @ V (32x64): 4 WMMAs accumulate into o[] ----
    v16bf ap = load_frag_a(pst, 32, lane);
    v16bf bv0 = load_frag_b(vbase + (size_t)0  * SLEN + k0, SLEN, lane);
    v16bf bv1 = load_frag_b(vbase + (size_t)16 * SLEN + k0, SLEN, lane);
    v16bf bv2 = load_frag_b(vbase + (size_t)32 * SLEN + k0, SLEN, lane);
    v16bf bv3 = load_frag_b(vbase + (size_t)48 * SLEN + k0, SLEN, lane);
    o[0] = WMMA_BF16(ap, bv0, o[0]);
    o[1] = WMMA_BF16(ap, bv1, o[1]);
    o[2] = WMMA_BF16(ap, bv2, o[2]);
    o[3] = WMMA_BF16(ap, bv3, o[3]);
  }

  // normalize and write ctx as [b, s, h*64+hd] bf16 for the projection GEMM
  const int mofs = (lane < 16) ? 0 : 8;
#pragma unroll
  for (int r = 0; r < 8; ++r) {
    int s = qt * 16 + r + mofs;
    float inv = 1.0f / lrow[r];
#pragma unroll
    for (int j = 0; j < 4; ++j) {
      int dcol = h * HDIM + j * 16 + (lane & 15);
      CTX[((size_t)b * SLEN + s) * DDIM + dcol] = f32_to_bf16(o[j][r] * inv);
    }
  }
}

// ---------------- output projection GEMM ----------------
// OUT[8192,1024] (f32) = CTX[8192,1024](bf16) @ Wout; WoT is [1024,1024] N-major.
// Same double-buffered async-LDS pipeline as the QKV GEMM.
__global__ __launch_bounds__(256) void proj_gemm_kernel(const __bf16* __restrict__ CTX,
                                                        const __bf16* __restrict__ WoT,
                                                        float* __restrict__ OUT) {
  __shared__ __align__(16) __bf16 Astage[2][128][64];
  __shared__ __align__(16) __bf16 Bstage[2][64][64];

  const int tid  = threadIdx.x;
  const int lane = tid & 31;
  const int wave = tid >> 5;
  const int n0   = blockIdx.x * 64;
  const int mblk = blockIdx.y * 128;
  const int m0   = mblk + wave * 16;

  v8f acc[4] = {vzero8(), vzero8(), vzero8(), vzero8()};

  stage_tiles(CTX, WoT, mblk, n0, 0, Astage[0], Bstage[0], tid);
  int cur = 0;
  for (int kc = 0; kc < DDIM; kc += 64) {
    const bool haveNext = (kc + 64) < DDIM;
    if (haveNext)
      stage_tiles(CTX, WoT, mblk, n0, kc + 64, Astage[cur ^ 1], Bstage[cur ^ 1], tid);
    wait_async(haveNext);
    __syncthreads();
    compute_chunk(Astage[cur], Bstage[cur], wave, lane, acc);
    __syncthreads();
    cur ^= 1;
  }

#pragma unroll
  for (int j = 0; j < 4; ++j) {
    int n = n0 + j * 16 + (lane & 15);
#pragma unroll
    for (int r = 0; r < 8; ++r) {
      int m = m0 + r + ((lane < 16) ? 0 : 8);
      OUT[(size_t)m * DDIM + n] = acc[j][r];
    }
  }
}

// ---------------- launch ----------------
extern "C" void kernel_launch(void* const* d_in, const int* in_sizes, int n_in,
                              void* d_out, int out_size, void* d_ws, size_t ws_size,
                              hipStream_t stream) {
  (void)in_sizes; (void)n_in; (void)out_size; (void)ws_size;
  const float* x     = (const float*)d_in[0];
  // d_in[1] = causal_mask (tril) — applied analytically, not read
  const float* Wqkv  = (const float*)d_in[2];
  const float* Wout  = (const float*)d_in[3];
  float* out = (float*)d_out;

  // workspace layout (bf16 elements); total ~88 MB
  __bf16* Xb  = (__bf16*)d_ws;
  __bf16* WqT = Xb  + (size_t)MROWS * DDIM;    // [3072,1024]
  __bf16* WoT = WqT + (size_t)NQKV * DDIM;     // [1024,1024]
  __bf16* Qb  = WoT + (size_t)DDIM * DDIM;     // [b,h,s,hd]
  __bf16* Kb  = Qb  + (size_t)MROWS * DDIM;
  __bf16* VTb = Kb  + (size_t)MROWS * DDIM;    // [b,h,hd,s]
  __bf16* CTX = VTb + (size_t)MROWS * DDIM;    // [b,s,d]

  const int nx = MROWS * DDIM;
  cast_f32_to_bf16<<<(nx + 255) / 256, 256, 0, stream>>>(x, Xb, nx);
  transpose_cast<<<(DDIM * NQKV + 255) / 256, 256, 0, stream>>>(Wqkv, WqT, DDIM, NQKV);
  transpose_cast<<<(DDIM * DDIM + 255) / 256, 256, 0, stream>>>(Wout, WoT, DDIM, DDIM);

  qkv_gemm_kernel<<<dim3(NQKV / 64, MROWS / 128), 256, 0, stream>>>(Xb, WqT, Qb, Kb, VTb);
  attn_kernel<<<dim3(SLEN / 128, BATCH * NH), 256, 0, stream>>>(Qb, Kb, VTb, CTX);
  proj_gemm_kernel<<<dim3(DDIM / 64, MROWS / 128), 256, 0, stream>>>(CTX, WoT, out);
}
